// SHAN_17497696764632
// MI455X (gfx1250) — compile-verified, matrix operand order checked by reading, not run on hardware
//
#include <hip/hip_runtime.h>
#include <hip/hip_bf16.h>

// ---------- types for WMMA ----------
typedef __attribute__((ext_vector_type(16))) __bf16 v16bf;
typedef __attribute__((ext_vector_type(8)))  __bf16 v8bf;
typedef __attribute__((ext_vector_type(8)))  float  v8f;

union BfFrag {
    v16bf v;
    v8bf  h[2];
};

__device__ __forceinline__ unsigned short f2bf(float f) {
    unsigned int u = __float_as_uint(f);
    u += 0x7FFFu + ((u >> 16) & 1u);   // round-to-nearest-even
    return (unsigned short)(u >> 16);
}

// ---------- kernel 1: f32 -> bf16 conversion of item_emb (vectorized x4) ----------
__global__ __launch_bounds__(256) void cvt_bf16_kernel(
    const float4* __restrict__ x, ushort4* __restrict__ y, int n4) {
    int i = blockIdx.x * blockDim.x + threadIdx.x;
    if (i < n4) {
        float4 v = x[i];
        ushort4 o;
        o.x = f2bf(v.x); o.y = f2bf(v.y); o.z = f2bf(v.z); o.w = f2bf(v.w);
        y[i] = o;
    }
}

// ---------- kernel 2: SHAN double attention pooling (one block per batch row) ----------
#define L_LONG 50
#define SHORTL 10

__global__ __launch_bounds__(64) void shan_pool_kernel(
    const int*   __restrict__ seq,
    const int*   __restrict__ user_id,
    const float* __restrict__ item_emb,
    const float* __restrict__ user_emb,
    const float* __restrict__ w1, const float* __restrict__ wb1, const float* __restrict__ b1,
    const float* __restrict__ w2, const float* __restrict__ wb2, const float* __restrict__ b2,
    unsigned short* __restrict__ pooled_bf)
{
    __shared__ float e_s[L_LONG][65];   // padded rows -> conflict-free column & row access
    __shared__ float Ws[64][64];        // broadcast reads only
    __shared__ float u_s[64];
    __shared__ float wb_s[64];
    __shared__ float s_s[64];
    __shared__ float red[2];
    __shared__ int   seq_s[L_LONG];

    const int b   = blockIdx.x;
    const int tid = threadIdx.x;        // 0..63

    if (tid < L_LONG) seq_s[tid] = seq[b * L_LONG + tid];
    __syncthreads();

    u_s[tid]  = user_emb[(size_t)user_id[b] * 64 + tid];
    wb_s[tid] = wb1[tid] + b1[tid];
    for (int l = 0; l < L_LONG; ++l)
        e_s[l][tid] = item_emb[(size_t)seq_s[l] * 64 + tid];
    for (int r = 0; r < 64; ++r)
        Ws[r][tid] = w1[r * 64 + tid];
    __syncthreads();

    // ---- pool 1: long-term attention over L=50 ----
    if (tid < L_LONG) {
        float acc = 0.f;
        for (int d = 0; d < 64; ++d) {
            float hh = wb_s[d];
            #pragma unroll 8
            for (int k = 0; k < 64; ++k) hh += e_s[tid][k] * Ws[d][k];
            acc += fmaxf(hh, 0.f) * u_s[d];
        }
        s_s[tid] = (seq_s[tid] == 0) ? -1000000000.0f : acc;
    }
    __syncthreads();
    if (tid == 0) {
        float m = -3.4e38f;
        for (int l = 0; l < L_LONG; ++l) m = fmaxf(m, s_s[l]);
        red[0] = m;
    }
    __syncthreads();
    if (tid < L_LONG) s_s[tid] = __expf(s_s[tid] - red[0]);
    __syncthreads();
    if (tid == 0) {
        float s = 0.f;
        for (int l = 0; l < L_LONG; ++l) s += s_s[l];
        red[1] = 1.f / s;
    }
    __syncthreads();
    float lt = 0.f;
    for (int l = 0; l < L_LONG; ++l) lt += s_s[l] * e_s[l][tid];
    lt *= red[1];
    __syncthreads();   // all reads of e_s / Ws done before overwrite

    // ---- build ls = [long_term ; e[:, -10:, :]] into rows 0..10 ----
    e_s[0][tid] = lt;
    for (int j = 0; j < SHORTL; ++j)
        e_s[1 + j][tid] = e_s[L_LONG - SHORTL + j][tid];
    wb_s[tid] = wb2[tid] + b2[tid];
    for (int r = 0; r < 64; ++r)
        Ws[r][tid] = w2[r * 64 + tid];
    __syncthreads();

    // ---- pool 2: long+short attention over L=11 ----
    const int L2 = SHORTL + 1;
    if (tid < L2) {
        float acc = 0.f;
        for (int d = 0; d < 64; ++d) {
            float hh = wb_s[d];
            #pragma unroll 8
            for (int k = 0; k < 64; ++k) hh += e_s[tid][k] * Ws[d][k];
            acc += fmaxf(hh, 0.f) * u_s[d];
        }
        bool msk = (tid > 0) && (seq_s[L_LONG - SHORTL + tid - 1] == 0);
        s_s[tid] = msk ? -1000000000.0f : acc;
    }
    __syncthreads();
    if (tid == 0) {
        float m = -3.4e38f;
        for (int l = 0; l < L2; ++l) m = fmaxf(m, s_s[l]);
        red[0] = m;
    }
    __syncthreads();
    if (tid < L2) s_s[tid] = __expf(s_s[tid] - red[0]);
    __syncthreads();
    if (tid == 0) {
        float s = 0.f;
        for (int l = 0; l < L2; ++l) s += s_s[l];
        red[1] = 1.f / s;
    }
    __syncthreads();
    float p = 0.f;
    for (int l = 0; l < L2; ++l) p += s_s[l] * e_s[l][tid];
    p *= red[1];

    pooled_bf[(size_t)b * 64 + tid] = f2bf(p);
}

// ---------- kernel 3: scores = pooled @ item_emb^T via bf16 WMMA ----------
// A: (Bb x 64) bf16 row-major, B: (NI x 64) bf16 row-major, C: (Bb x NI) f32.
// One wave computes four 16x16 tiles (reusing its A fragments), 2x
// v_wmma_f32_16x16x32_bf16 per tile (K=64).
//
// Aligned variant: NI is a compile-time constant (and Bb,NI multiples of 16),
// so the 8 row stores per tile collapse to global_store_b32 with immediate
// offsets (j*NI_C*4 < 2^23) off one base address, NT temporal hint, and the
// only branch is a wave-uniform scalar residual check.
template <int NI_C>
__global__ __launch_bounds__(256) void gemm_wmma_aligned_kernel(
    const unsigned short* __restrict__ Abf,
    const unsigned short* __restrict__ Bbf,
    float* __restrict__ C,
    int nTiles)
{
    const int wave = __builtin_amdgcn_readfirstlane(threadIdx.x >> 5); // SGPR -> scalar branches
    const int lane = threadIdx.x & 31;
    const int m0 = blockIdx.y << 4;
    const int ntBase = (blockIdx.x * 8 + wave) * 4;
    if (ntBase >= nTiles) return;

    const int row  = lane & 15;
    // Lanes 0-15 hold K {0..7,16..23}; lanes 16-31 hold K {8..15,24..31}
    const int koff = (lane & 16) ? 8 : 0;

    const __bf16* A  = (const __bf16*)Abf;
    const __bf16* Bm = (const __bf16*)Bbf;

    const __bf16* ap = A + (size_t)(m0 + row) * 64 + koff;
    BfFrag a0, a1;
    a0.h[0] = *(const v8bf*)(ap);
    a0.h[1] = *(const v8bf*)(ap + 16);
    a1.h[0] = *(const v8bf*)(ap + 32);
    a1.h[1] = *(const v8bf*)(ap + 48);

    const int mBase   = m0 + ((lane & 16) >> 1);   // +8 for upper half-wave
    const int colLane = lane & 15;

    #pragma unroll
    for (int t = 0; t < 4; ++t) {
        const int nt = ntBase + t;
        if (nt >= nTiles) break;                   // wave-uniform scalar branch
        const int n0 = nt << 4;

        // B operand: lane n holds column n of (K x N) B = row (n0+n) of item_emb
        const __bf16* bp = Bm + (size_t)(n0 + row) * 64 + koff;
        BfFrag b0, b1;
        b0.h[0] = *(const v8bf*)(bp);
        b0.h[1] = *(const v8bf*)(bp + 16);
        b1.h[0] = *(const v8bf*)(bp + 32);
        b1.h[1] = *(const v8bf*)(bp + 48);

        v8f c = {};
        c = __builtin_amdgcn_wmma_f32_16x16x32_bf16(false, a0.v, false, b0.v,
                                                    (short)0, c, false, false);
        c = __builtin_amdgcn_wmma_f32_16x16x32_bf16(false, a1.v, false, b1.v,
                                                    (short)0, c, false, false);

        // C layout: VGPR j -> M = j (lanes 0-15) / j+8 (lanes 16-31), N = lane%16
        float* p = C + (size_t)mBase * NI_C + (n0 + colLane);
        #pragma unroll
        for (int j = 0; j < 8; ++j)
            __builtin_nontemporal_store(c[j], p + (size_t)j * NI_C); // imm offsets
    }
}

// Generic fallback (runtime NI, guarded edges)
__global__ __launch_bounds__(256) void gemm_wmma_kernel(
    const unsigned short* __restrict__ Abf,
    const unsigned short* __restrict__ Bbf,
    float* __restrict__ C,
    int Bb, int NI, int nTiles)
{
    const int wave = __builtin_amdgcn_readfirstlane(threadIdx.x >> 5);
    const int lane = threadIdx.x & 31;
    const int m0 = blockIdx.y << 4;
    const int ntBase = (blockIdx.x * 8 + wave) * 4;
    if (ntBase >= nTiles) return;

    const int row  = lane & 15;
    const int koff = (lane & 16) ? 8 : 0;

    const __bf16* A  = (const __bf16*)Abf;
    const __bf16* Bm = (const __bf16*)Bbf;

    int arow = m0 + row; if (arow >= Bb) arow = Bb - 1;
    const __bf16* ap = A + (size_t)arow * 64 + koff;
    BfFrag a0, a1;
    a0.h[0] = *(const v8bf*)(ap);
    a0.h[1] = *(const v8bf*)(ap + 16);
    a1.h[0] = *(const v8bf*)(ap + 32);
    a1.h[1] = *(const v8bf*)(ap + 48);

    const int mBase   = m0 + ((lane & 16) >> 1);
    const int colLane = lane & 15;

    for (int t = 0; t < 4; ++t) {
        const int nt = ntBase + t;
        if (nt >= nTiles) break;
        const int n0 = nt << 4;

        int brow = n0 + row; if (brow >= NI) brow = NI - 1;
        const __bf16* bp = Bm + (size_t)brow * 64 + koff;
        BfFrag b0, b1;
        b0.h[0] = *(const v8bf*)(bp);
        b0.h[1] = *(const v8bf*)(bp + 16);
        b1.h[0] = *(const v8bf*)(bp + 32);
        b1.h[1] = *(const v8bf*)(bp + 48);

        v8f c = {};
        c = __builtin_amdgcn_wmma_f32_16x16x32_bf16(false, a0.v, false, b0.v,
                                                    (short)0, c, false, false);
        c = __builtin_amdgcn_wmma_f32_16x16x32_bf16(false, a1.v, false, b1.v,
                                                    (short)0, c, false, false);

        const int col = n0 + colLane;
        float* p = C + (size_t)mBase * NI + col;
        #pragma unroll
        for (int j = 0; j < 8; ++j) {
            if (col < NI && (mBase + j) < Bb)
                __builtin_nontemporal_store(c[j], p);
            p += NI;
        }
    }
}

extern "C" void kernel_launch(void* const* d_in, const int* in_sizes, int n_in,
                              void* d_out, int out_size, void* d_ws, size_t ws_size,
                              hipStream_t stream) {
    const int*   seq      = (const int*)d_in[0];
    const int*   user_id  = (const int*)d_in[1];
    const float* item_emb = (const float*)d_in[2];
    const float* user_emb = (const float*)d_in[3];
    const float* long_w        = (const float*)d_in[4];
    const float* long_w_bias   = (const float*)d_in[5];
    const float* long_b        = (const float*)d_in[6];
    const float* ls_w          = (const float*)d_in[7];
    const float* ls_w_bias     = (const float*)d_in[8];
    const float* ls_b          = (const float*)d_in[9];
    float* scores = (float*)d_out;

    const int Bb = in_sizes[1];            // 2048
    const int D  = 64;
    const int NI = in_sizes[2] / D;        // 100000

    // workspace: [ item_emb bf16 : NI*D ][ pooled bf16 : Bb*D ]
    unsigned short* itemb_bf  = (unsigned short*)d_ws;
    unsigned short* pooled_bf = itemb_bf + (size_t)NI * D;

    // 1) convert item_emb to bf16 (D=64 -> total divisible by 4)
    {
        int n4 = (NI * D) / 4;
        cvt_bf16_kernel<<<(n4 + 255) / 256, 256, 0, stream>>>(
            (const float4*)item_emb, (ushort4*)itemb_bf, n4);
    }

    // 2) double attention pooling -> pooled (bf16)
    shan_pool_kernel<<<Bb, 64, 0, stream>>>(
        seq, user_id, item_emb, user_emb,
        long_w, long_w_bias, long_b,
        ls_w, ls_w_bias, ls_b,
        pooled_bf);

    // 3) scores = pooled @ item_emb^T  (WMMA bf16 -> f32)
    {
        const int nTiles = (NI + 15) / 16;             // 6250
        const int wavesNeeded = (nTiles + 3) / 4;      // 4 N-tiles per wave
        if (NI == 100000 && (Bb & 15) == 0) {
            dim3 grid((wavesNeeded + 7) / 8, Bb / 16);
            gemm_wmma_aligned_kernel<100000><<<grid, 256, 0, stream>>>(
                pooled_bf, itemb_bf, scores, nTiles);
        } else {
            dim3 grid((wavesNeeded + 7) / 8, (Bb + 15) / 16);
            gemm_wmma_kernel<<<grid, 256, 0, stream>>>(pooled_bf, itemb_bf, scores,
                                                       Bb, NI, nTiles);
        }
    }
}
